// SimpleGCNModel_1683627180174
// MI455X (gfx1250) — matrix operand order repaired and checked by reading, not compiled, not run on hardware
//
#include <hip/hip_runtime.h>
#include <hip/hip_bf16.h>

// ---------- types for WMMA ----------
typedef __attribute__((ext_vector_type(16))) __bf16          v16bf;
typedef __attribute__((ext_vector_type(8)))  float           v8f;
typedef __attribute__((ext_vector_type(16))) unsigned short  v16u;
typedef __attribute__((ext_vector_type(8)))  unsigned short  v8u;

union ABFrag { v16u us; v16bf bf; struct { v8u lo, hi; } h; };

// ---------- helpers ----------
__device__ __forceinline__ unsigned short f2bf(float f) {
    unsigned int u = __float_as_uint(f);
    unsigned int r = (u + 0x7FFFu + ((u >> 16) & 1u)) >> 16;   // RNE
    return (unsigned short)r;
}

// ---------- elementwise kernels ----------
__global__ void k_fill(float* p, float v, int n) {
    int i = blockIdx.x * blockDim.x + threadIdx.x;
    if (i < n) p[i] = v;
}

__global__ void k_edge_deg(const int* __restrict__ dst, float* __restrict__ deg, int E) {
    int e = blockIdx.x * blockDim.x + threadIdx.x;
    if (e < E) atomicAdd(&deg[dst[e]], 1.0f);
}

__global__ void k_rsqrt(const float* __restrict__ deg, float* __restrict__ dinv, int n) {
    int i = blockIdx.x * blockDim.x + threadIdx.x;
    if (i < n) {
        float d = deg[i];
        dinv[i] = d > 0.0f ? rsqrtf(d) : 0.0f;
    }
}

__global__ void k_f2bf(const float* __restrict__ in, unsigned short* __restrict__ out, long long n) {
    long long i = (long long)blockIdx.x * blockDim.x + threadIdx.x;
    if (i < n) out[i] = f2bf(in[i]);
}

// ---------- pack W[K,OUT] (f32 row-major) into WMMA B-fragment order (bf16) ----------
// Fragment: for (colTile, kTile) 16x32 chunk, lane = 16*half + (c&15) holds
// contiguous elements i=0..15 with k = 32*kTile + 16*half + i.
// packed[ ((colTile*numKT + kTile)*32 + lane)*16 + i ]
__global__ void k_packW(const float* __restrict__ W, unsigned short* __restrict__ packed,
                        int K, int OUT) {
    int idx = blockIdx.x * blockDim.x + threadIdx.x;
    if (idx >= K * OUT) return;
    int k = idx / OUT;
    int c = idx - k * OUT;
    int numKT   = K >> 5;
    int kTile   = k >> 5;
    int half    = (k >> 4) & 1;
    int i       = k & 15;
    int colTile = c >> 4;
    int lane    = (half << 4) | (c & 15);
    size_t off = ((size_t)(colTile * numKT + kTile) * 32 + lane) * 16 + i;
    packed[off] = f2bf(W[idx]);
}

// ---------- WMMA GEMM: H[Npad,OUT] = A[Npad,K](bf16) @ W(packed bf16) ----------
// Compile-time K/OUT: fully unrolled k-loop, immediate-offset b128 loads.
// One wave -> 16-row x 64-col strip (4 col tiles); A rows padded to 16 with
// zeros so no predication anywhere in the hot path (EXEC stays all-ones).
template <int K, int OUT>
__global__ __launch_bounds__(256)
void k_gemm_wmma(const unsigned short* __restrict__ A,
                 const unsigned short* __restrict__ Wp,
                 float* __restrict__ H,
                 int totalTiles) {
    constexpr int numKT    = K >> 5;
    constexpr int colQuads = OUT >> 6;

    const int lane = threadIdx.x & 31;
    const int wave = blockIdx.x * (blockDim.x >> 5) + (threadIdx.x >> 5);
    if (wave >= totalTiles) return;                 // wave-uniform exit

    const int quad    = wave % colQuads;
    const int rowTile = wave / colQuads;
    const int half = lane >> 4;
    const int l16  = lane & 15;

    const unsigned short* pa = A + (size_t)(rowTile * 16 + l16) * K + 16 * half;
    const unsigned short* pb0 = Wp + ((size_t)(quad * 4 * numKT) * 32 + lane) * 16;

    v8f acc[4] = {};
#pragma unroll
    for (int kt = 0; kt < numKT; ++kt) {
        ABFrag a;
        a.h.lo = *(const v8u*)(pa + kt * 32);
        a.h.hi = *(const v8u*)(pa + kt * 32 + 8);
#pragma unroll
        for (int q = 0; q < 4; ++q) {
            const unsigned short* pb = pb0 + (size_t)(q * numKT + kt) * 512;
            ABFrag b;
            b.h.lo = *(const v8u*)(pb);
            b.h.hi = *(const v8u*)(pb + 8);
            acc[q] = __builtin_amdgcn_wmma_f32_16x16x32_bf16(
                false, a.bf, false, b.bf, (short)0, acc[q], false, false);
        }
    }

    // D layout: VGPR r -> row = rowTile*16 + 8*half + r, col = quad*64 + q*16 + l16
    float* ph = H + (size_t)(rowTile * 16 + 8 * half) * OUT + quad * 64 + l16;
#pragma unroll
    for (int q = 0; q < 4; ++q)
#pragma unroll
        for (int r = 0; r < 8; ++r)
            ph[(size_t)r * OUT + q * 16] = acc[q][r];
}

// ---------- edge scatter: agg[dst] += h[src] * dinv[src]*dinv[dst] ----------
__global__ void k_scatter(const int* __restrict__ src, const int* __restrict__ dst,
                          const float* __restrict__ dinv,
                          const float* __restrict__ h, float* __restrict__ agg,
                          int E, int OUT) {
    int idx = blockIdx.x * blockDim.x + threadIdx.x;
    int groups = OUT >> 2;
    int e = idx / groups;
    int c4 = idx - e * groups;
    if (e >= E) return;
    int s = src[e], d = dst[e];
    float nrm = dinv[s] * dinv[d];
    const float4 v = *(const float4*)(h + (size_t)s * OUT + 4 * c4);
    float* ad = agg + (size_t)d * OUT + 4 * c4;
    atomicAdd(ad + 0, v.x * nrm);
    atomicAdd(ad + 1, v.y * nrm);
    atomicAdd(ad + 2, v.z * nrm);
    atomicAdd(ad + 3, v.w * nrm);
}

// ---------- epilogue: self-loop + bias + optional relu; also emit bf16 copy ----------
__global__ void k_bias_relu(float* __restrict__ agg, const float* __restrict__ h,
                            const float* __restrict__ dinv, const float* __restrict__ b,
                            unsigned short* __restrict__ out_bf,
                            int N, int OUT, int doRelu) {
    int idx = blockIdx.x * blockDim.x + threadIdx.x;
    if (idx >= N * OUT) return;
    int i = idx / OUT;
    int c = idx - i * OUT;
    float di = dinv[i];
    float v = agg[idx] + h[idx] * di * di + b[c];
    if (doRelu) v = fmaxf(v, 0.0f);
    agg[idx] = v;
    out_bf[idx] = f2bf(v);                 // next layer's WMMA A-matrix input
}

// ---------- pooling ----------
__global__ void k_pool(const float* __restrict__ h, const int* __restrict__ batch,
                       float* __restrict__ pooled, int N, int OUT) {
    int idx = blockIdx.x * blockDim.x + threadIdx.x;
    if (idx >= N * OUT) return;
    int i = idx / OUT;
    int c = idx - i * OUT;
    atomicAdd(&pooled[(size_t)batch[i] * OUT + c], h[idx]);
}

__global__ void k_cnt(const int* __restrict__ batch, float* __restrict__ cnt, int N) {
    int i = blockIdx.x * blockDim.x + threadIdx.x;
    if (i < N) atomicAdd(&cnt[batch[i]], 1.0f);
}

// ---------- tiny classifier: out[G,C] = (pooled/cnt) @ Wl + bl ----------
__global__ void k_final(const float* __restrict__ pooled, const float* __restrict__ cnt,
                        const float* __restrict__ Wl, const float* __restrict__ bl,
                        float* __restrict__ out, int G, int C, int H) {
    int idx = blockIdx.x * blockDim.x + threadIdx.x;
    if (idx >= G * C) return;
    int g = idx / C;
    int c = idx - g * C;
    float inv = 1.0f / fmaxf(cnt[g], 1.0f);
    const float* pg = pooled + (size_t)g * H;
    float s = 0.0f;
    for (int k = 0; k < H; ++k) s = fmaf(pg[k] * inv, Wl[(size_t)k * C + c], s);
    out[idx] = s + bl[c];
}

static inline int nblk(long long n) { return (int)((n + 255) / 256); }

extern "C" void kernel_launch(void* const* d_in, const int* in_sizes, int n_in,
                              void* d_out, int out_size, void* d_ws, size_t ws_size,
                              hipStream_t stream) {
    const float* x    = (const float*)d_in[0];
    const int*   ei   = (const int*)d_in[1];
    const int*   batch= (const int*)d_in[2];
    const float* W1   = (const float*)d_in[3];
    const float* b1   = (const float*)d_in[4];
    const float* W2   = (const float*)d_in[5];
    const float* b2   = (const float*)d_in[6];
    const float* W3   = (const float*)d_in[7];
    const float* b3   = (const float*)d_in[8];
    const float* Wl   = (const float*)d_in[9];
    const float* bl   = (const float*)d_in[10];
    float* out = (float*)d_out;

    const int F  = 32;
    const int N  = in_sizes[0] / F;
    const int E  = in_sizes[1] / 2;
    const int H1 = in_sizes[4];          // 64
    const int H2 = in_sizes[6];          // 128
    const int C  = in_sizes[10];         // 8
    const int G  = out_size / C;         // 128
    const int Npad = ((N + 15) / 16) * 16;
    const int rowTiles = Npad / 16;

    const int* srcp = ei;
    const int* dstp = ei + E;

    // ---- workspace layout ----
    char* p = (char*)d_ws;
    auto alloc = [&](size_t bytes) { void* r = (void*)p; p += (bytes + 255) & ~(size_t)255; return r; };
    float*          deg    = (float*)alloc((size_t)N * 4);
    float*          dinv   = (float*)alloc((size_t)N * 4);
    unsigned short* bufA   = (unsigned short*)alloc((size_t)Npad * H2 * 2); // bf16 activations (row-padded)
    float*          bufH   = (float*)alloc((size_t)Npad * H2 * 4);          // GEMM output (row-padded)
    float*          bufB   = (float*)alloc((size_t)N * H2 * 4);             // aggregated output
    unsigned short* wb     = (unsigned short*)alloc((size_t)H2 * H2 * 2);   // packed bf16 weights
    float*          pooled = (float*)alloc((size_t)G * H2 * 4);
    float*          cnt    = (float*)alloc((size_t)G * 4);
    (void)ws_size; (void)n_in;

    // zero bufA so padded tail rows feed zeros to WMMA on every layer
    hipMemsetAsync(bufA, 0, (size_t)Npad * H2 * 2, stream);

    // ---- degrees (with self-loops), shared across layers ----
    k_fill<<<nblk(N), 256, 0, stream>>>(deg, 1.0f, N);
    k_edge_deg<<<nblk(E), 256, 0, stream>>>(dstp, deg, E);
    k_rsqrt<<<nblk(N), 256, 0, stream>>>(deg, dinv, N);

    // layer 1 input: convert x once; layers 2/3 consume bf16 emitted by epilogue
    k_f2bf<<<nblk((long long)N * F), 256, 0, stream>>>(x, bufA, (long long)N * F);

    auto post = [&](int OUT, const float* b, int relu) {
        hipMemsetAsync(bufB, 0, (size_t)N * OUT * 4, stream);
        long long work = (long long)E * (OUT / 4);
        k_scatter<<<nblk(work), 256, 0, stream>>>(srcp, dstp, dinv, bufH, bufB, E, OUT);
        k_bias_relu<<<nblk((long long)N * OUT), 256, 0, stream>>>(bufB, bufH, dinv, b, bufA, N, OUT, relu);
    };

    // ---- layer 1: [N,32] @ [32,64] ----
    {
        k_packW<<<nblk(32 * 64), 256, 0, stream>>>(W1, wb, 32, 64);
        int tiles = rowTiles * (64 / 64);
        k_gemm_wmma<32, 64><<<(tiles + 7) / 8, 256, 0, stream>>>(bufA, wb, bufH, tiles);
        post(64, b1, 1);
    }
    // ---- layer 2: [N,64] @ [64,128] ----
    {
        k_packW<<<nblk(64 * 128), 256, 0, stream>>>(W2, wb, 64, 128);
        int tiles = rowTiles * (128 / 64);
        k_gemm_wmma<64, 128><<<(tiles + 7) / 8, 256, 0, stream>>>(bufA, wb, bufH, tiles);
        post(128, b2, 1);
    }
    // ---- layer 3: [N,128] @ [128,128] ----
    {
        k_packW<<<nblk(128 * 128), 256, 0, stream>>>(W3, wb, 128, 128);
        int tiles = rowTiles * (128 / 64);
        k_gemm_wmma<128, 128><<<(tiles + 7) / 8, 256, 0, stream>>>(bufA, wb, bufH, tiles);
        post(128, b3, 0);
    }

    // ---- global mean pool + classifier ----
    hipMemsetAsync(pooled, 0, (size_t)G * H2 * 4, stream);
    hipMemsetAsync(cnt,    0, (size_t)G * 4,      stream);
    k_pool<<<nblk((long long)N * H2), 256, 0, stream>>>(bufB, batch, pooled, N, H2);
    k_cnt<<<nblk(N), 256, 0, stream>>>(batch, cnt, N);
    k_final<<<nblk((long long)G * C), 256, 0, stream>>>(pooled, cnt, Wl, bl, out, G, C, H2);
}
